// NodeV1Model_28484223107667
// MI455X (gfx1250) — matrix operand order repaired and checked by reading, not compiled, not run on hardware
//
#include <hip/hip_runtime.h>
#include <hip/hip_bf16.h>

// ---------------------------------------------------------------------------
// GraphNet node block for MI455X (gfx1250), bf16 WMMA (v_wmma_f32_16x16x32_bf16)
// M=32 per wave (two 16-row tiles) so every B fragment load feeds 2 WMMAs.
// ---------------------------------------------------------------------------

typedef __attribute__((ext_vector_type(16))) __bf16 v16bf;
typedef __attribute__((ext_vector_type(8)))  __bf16 bf16x8;
typedef __attribute__((ext_vector_type(8)))  float  v8f;

#define LDSTRIDE 136   // 16x128 bf16 tile padded to 136 (272B rows -> bank spread)

// Load a pre-swizzled B fragment: 16 consecutive bf16 per lane (2x b128 loads).
__device__ __forceinline__ v16bf load_bfrag(const __bf16* __restrict__ p) {
  bf16x8 lo = *(const bf16x8*)(p);
  bf16x8 hi = *(const bf16x8*)(p + 8);
  v16bf r;
#pragma unroll
  for (int i = 0; i < 8; ++i) { r[i] = lo[i]; r[i + 8] = hi[i]; }
  return r;
}

// Write 8 consecutive fp32 (2x float4 loads) into A-fragment elements [at, at+8).
__device__ __forceinline__ void put8(v16bf& f, int at, const float* __restrict__ p, float s) {
  float4 a = *(const float4*)(p);
  float4 b = *(const float4*)(p + 4);
  f[at + 0] = (__bf16)(a.x * s); f[at + 1] = (__bf16)(a.y * s);
  f[at + 2] = (__bf16)(a.z * s); f[at + 3] = (__bf16)(a.w * s);
  f[at + 4] = (__bf16)(b.x * s); f[at + 5] = (__bf16)(b.y * s);
  f[at + 6] = (__bf16)(b.z * s); f[at + 7] = (__bf16)(b.w * s);
}

// A fragment (16x32 bf16) from an LDS row-major tile, chunk kc (2x ds_load_b128).
__device__ __forceinline__ v16bf load_a_lds(const __bf16* __restrict__ hrow,
                                            int m, int hl, int kc) {
  const __bf16* p = hrow + m * LDSTRIDE + kc * 32;
  bf16x8 lo = *(const bf16x8*)(p + hl * 8);
  bf16x8 hi = *(const bf16x8*)(p + 16 + hl * 8);
  v16bf r;
#pragma unroll
  for (int i = 0; i < 8; ++i) { r[i] = lo[i]; r[i + 8] = hi[i]; }
  return r;
}

__device__ __forceinline__ v8f wmma_bf16(v16bf a, v16bf b, v8f c) {
  return __builtin_amdgcn_wmma_f32_16x16x32_bf16(false, a, false, b, (short)0, c,
                                                 false, false);
}

// ---------------------------------------------------------------------------
// Workspace prep kernels
// ---------------------------------------------------------------------------

__global__ void k_zero(float* __restrict__ p, int n) {
  int i = blockIdx.x * 256 + threadIdx.x;
  if (i < n) p[i] = 0.f;
}

// Swizzle fp32 weight W[K][Ncols] (row-major) into bf16 B-fragment blocks:
// block (kc, nt) of 512 elems; elem (lane, j): K = kc*32 + (lane<16?0:8) + (j<8?j:j+8),
// N = nt*16 + (lane&15).  total = K*Ncols.
__global__ void k_swz(const float* __restrict__ W, __bf16* __restrict__ out,
                      int Ncols, int Nt, int total) {
  int idx = blockIdx.x * 256 + threadIdx.x;
  if (idx >= total) return;
  int blk  = idx >> 9;
  int win  = idx & 511;
  int lane = win >> 4;
  int j    = win & 15;
  int kc   = blk / Nt;
  int nt   = blk - kc * Nt;
  int k    = kc * 32 + ((lane & 16) ? 8 : 0) + ((j < 8) ? j : j + 8);
  int nn   = nt * 16 + (lane & 15);
  out[idx] = (__bf16)W[(size_t)k * Ncols + nn];
}

// ---------------------------------------------------------------------------
// Edge kernel: msg = (relu([x[row]|ea] @ mw1 + mb1) @ mw2 + mb2) * wts
// scatter-add into sums/cnt.  One 32-edge tile per wave, 8 waves per block.
// ---------------------------------------------------------------------------

__global__ __launch_bounds__(256) void k_edge(
    const float* __restrict__ x, const int* __restrict__ erow,
    const int* __restrict__ ecol, const float* __restrict__ ea,
    const float* __restrict__ wts, const float* __restrict__ mb1,
    const float* __restrict__ mb2, const __bf16* __restrict__ w1s,
    const __bf16* __restrict__ w2s, float* __restrict__ sums,
    float* __restrict__ cnt, int E, int ntiles) {
  __shared__ __bf16 lds_h[8][2][16 * LDSTRIDE];
  int wave = threadIdx.x >> 5;
  int tile = blockIdx.x * 8 + wave;
  if (tile >= ntiles) return;                 // wave-uniform: EXEC stays full
  int lane = threadIdx.x & 31;
  int m    = lane & 15;
  int hl   = lane >> 4;
  int n    = m;
  int ebase = tile * 32;
  int b0 = hl * 8;
  int b1 = 16 + hl * 8;

  // ---- gather A (2 tiles x 16 edges x 96 feats) into WMMA A-fragments ----
  v16bf A[2][3];
#pragma unroll
  for (int t = 0; t < 2; ++t) {
    int eg = ebase + t * 16 + m;
    eg = (eg < E) ? eg : (E - 1);             // branchless clamp
    const float* xr = x  + (size_t)erow[eg] * 64;
    const float* ar = ea + (size_t)eg * 32;
    put8(A[t][0], 0, xr + b0, 1.f);       put8(A[t][0], 8, xr + b1, 1.f);
    put8(A[t][1], 0, xr + 32 + b0, 1.f);  put8(A[t][1], 8, xr + 32 + b1, 1.f);
    put8(A[t][2], 0, ar + b0, 1.f);       put8(A[t][2], 8, ar + b1, 1.f);
  }

  // ---- GEMM1: h = relu(A(32x96) @ W1(96x128) + b1) -> LDS (bf16) ----
  const __bf16* w1p = w1s + lane * 16;
#pragma unroll
  for (int nt = 0; nt < 8; ++nt) {
    v8f acc0 = {};                            // SRC2 folds to inline 0
    v8f acc1 = {};
#pragma unroll
    for (int kc = 0; kc < 3; ++kc) {
      v16bf b = load_bfrag(w1p + ((size_t)(kc * 8 + nt) << 9));
      acc0 = wmma_bf16(A[0][kc], b, acc0);
      acc1 = wmma_bf16(A[1][kc], b, acc1);
    }
    float bias = mb1[nt * 16 + n];
#pragma unroll
    for (int r = 0; r < 8; ++r) {
      float v0 = acc0[r] + bias; v0 = v0 > 0.f ? v0 : 0.f;
      float v1 = acc1[r] + bias; v1 = v1 > 0.f ? v1 : 0.f;
      lds_h[wave][0][(r + hl * 8) * LDSTRIDE + nt * 16 + n] = (__bf16)v0;
      lds_h[wave][1][(r + hl * 8) * LDSTRIDE + nt * 16 + n] = (__bf16)v1;
    }
  }

  // per-lane destination/weight info for the 8 C-rows this lane owns, per tile
  int   dst[2][8];
  float ew[2][8];
#pragma unroll
  for (int t = 0; t < 2; ++t) {
#pragma unroll
    for (int r = 0; r < 8; ++r) {
      int e  = ebase + t * 16 + r + hl * 8;
      int e2 = (e < E) ? e : (E - 1);
      dst[t][r] = ecol[e2];
      ew[t][r]  = (e < E) ? wts[e2] : 0.f;    // weight-0 kills OOB contributions
    }
  }

  // ---- GEMM2: msg = (h @ W2(128x64) + b2) * wts ; scatter-add ----
  const __bf16* w2p = w2s + lane * 16;
#pragma unroll
  for (int nt2 = 0; nt2 < 4; ++nt2) {
    v8f acc0 = {};
    v8f acc1 = {};
#pragma unroll
    for (int kc = 0; kc < 4; ++kc) {
      v16bf b2 = load_bfrag(w2p + ((size_t)(kc * 4 + nt2) << 9));
      acc0 = wmma_bf16(load_a_lds(lds_h[wave][0], m, hl, kc), b2, acc0);
      acc1 = wmma_bf16(load_a_lds(lds_h[wave][1], m, hl, kc), b2, acc1);
    }
    float bias = mb2[nt2 * 16 + n];
#pragma unroll
    for (int r = 0; r < 8; ++r) {
      atomicAdd(&sums[(size_t)dst[0][r] * 64 + nt2 * 16 + n], (acc0[r] + bias) * ew[0][r]);
      atomicAdd(&sums[(size_t)dst[1][r] * 64 + nt2 * 16 + n], (acc1[r] + bias) * ew[1][r]);
    }
  }

  // incoming-edge counts (one lane column per row; lanes 0 and 16 cover 16 rows)
  if (n == 0) {
#pragma unroll
    for (int t = 0; t < 2; ++t) {
#pragma unroll
      for (int r = 0; r < 8; ++r) {
        int e = ebase + t * 16 + r + hl * 8;
        if (e < E) atomicAdd(&cnt[dst[t][r]], 1.0f);
      }
    }
  }
}

// ---------------------------------------------------------------------------
// Node kernel: out = relu([x | sums/max(cnt,1) | u[batch]] @ uw1 + ub1) @ uw2 + ub2
// ---------------------------------------------------------------------------

__global__ __launch_bounds__(256) void k_node(
    const float* __restrict__ x, const int* __restrict__ nbatch,
    const float* __restrict__ u, const float* __restrict__ sums,
    const float* __restrict__ cnt, const float* __restrict__ ub1,
    const float* __restrict__ ub2, const __bf16* __restrict__ u1s,
    const __bf16* __restrict__ u2s, float* __restrict__ out, int N, int ntiles) {
  __shared__ __bf16 lds_h[8][2][16 * LDSTRIDE];
  int wave = threadIdx.x >> 5;
  int tile = blockIdx.x * 8 + wave;
  if (tile >= ntiles) return;
  int lane = threadIdx.x & 31;
  int m    = lane & 15;
  int hl   = lane >> 4;
  int n    = m;
  int nbase = tile * 32;
  int b0 = hl * 8;
  int b1 = 16 + hl * 8;

  v16bf A[2][5];                              // K = 160 = 5 chunks of 32
#pragma unroll
  for (int t = 0; t < 2; ++t) {
    int ng = nbase + t * 16 + m;
    ng = (ng < N) ? ng : (N - 1);
    const float* xr = x    + (size_t)ng * 64;
    const float* sr = sums + (size_t)ng * 64;
    float invc = 1.0f / fmaxf(cnt[ng], 1.0f);
    const float* ur = u + (size_t)nbatch[ng] * 32;
    put8(A[t][0], 0, xr + b0, 1.f);        put8(A[t][0], 8, xr + b1, 1.f);
    put8(A[t][1], 0, xr + 32 + b0, 1.f);   put8(A[t][1], 8, xr + 32 + b1, 1.f);
    put8(A[t][2], 0, sr + b0, invc);       put8(A[t][2], 8, sr + b1, invc);
    put8(A[t][3], 0, sr + 32 + b0, invc);  put8(A[t][3], 8, sr + 32 + b1, invc);
    put8(A[t][4], 0, ur + b0, 1.f);        put8(A[t][4], 8, ur + b1, 1.f);
  }

  const __bf16* u1p = u1s + lane * 16;
#pragma unroll
  for (int nt = 0; nt < 8; ++nt) {
    v8f acc0 = {};
    v8f acc1 = {};
#pragma unroll
    for (int kc = 0; kc < 5; ++kc) {
      v16bf b = load_bfrag(u1p + ((size_t)(kc * 8 + nt) << 9));
      acc0 = wmma_bf16(A[0][kc], b, acc0);
      acc1 = wmma_bf16(A[1][kc], b, acc1);
    }
    float bias = ub1[nt * 16 + n];
#pragma unroll
    for (int r = 0; r < 8; ++r) {
      float v0 = acc0[r] + bias; v0 = v0 > 0.f ? v0 : 0.f;
      float v1 = acc1[r] + bias; v1 = v1 > 0.f ? v1 : 0.f;
      lds_h[wave][0][(r + hl * 8) * LDSTRIDE + nt * 16 + n] = (__bf16)v0;
      lds_h[wave][1][(r + hl * 8) * LDSTRIDE + nt * 16 + n] = (__bf16)v1;
    }
  }

  const __bf16* u2p = u2s + lane * 16;
#pragma unroll
  for (int nt2 = 0; nt2 < 4; ++nt2) {
    v8f acc0 = {};
    v8f acc1 = {};
#pragma unroll
    for (int kc = 0; kc < 4; ++kc) {
      v16bf b2 = load_bfrag(u2p + ((size_t)(kc * 4 + nt2) << 9));
      acc0 = wmma_bf16(load_a_lds(lds_h[wave][0], m, hl, kc), b2, acc0);
      acc1 = wmma_bf16(load_a_lds(lds_h[wave][1], m, hl, kc), b2, acc1);
    }
    float bias = ub2[nt2 * 16 + n];
#pragma unroll
    for (int r = 0; r < 8; ++r) {
      int nr0 = nbase + r + hl * 8;
      int nr1 = nbase + 16 + r + hl * 8;
      if (nr0 < N) out[(size_t)nr0 * 64 + nt2 * 16 + n] = acc0[r] + bias;
      if (nr1 < N) out[(size_t)nr1 * 64 + nt2 * 16 + n] = acc1[r] + bias;
    }
  }
}

// ---------------------------------------------------------------------------

extern "C" void kernel_launch(void* const* d_in, const int* in_sizes, int n_in,
                              void* d_out, int out_size, void* d_ws, size_t ws_size,
                              hipStream_t stream) {
  const float* x    = (const float*)d_in[0];
  const int*   ei   = (const int*)d_in[1];
  const float* ea   = (const float*)d_in[2];
  const float* u    = (const float*)d_in[3];
  const int*   nb   = (const int*)d_in[4];
  const float* wts  = (const float*)d_in[5];
  const float* mw1  = (const float*)d_in[6];
  const float* mb1  = (const float*)d_in[7];
  const float* mw2  = (const float*)d_in[8];
  const float* mb2  = (const float*)d_in[9];
  const float* uw1  = (const float*)d_in[10];
  const float* ub1  = (const float*)d_in[11];
  const float* uw2  = (const float*)d_in[12];
  const float* ub2  = (const float*)d_in[13];

  const int N = in_sizes[4];       // node_batch has N elements
  const int E = in_sizes[5];       // wts has E elements

  // workspace layout: sums[N*64] f32 | cnt[N] f32 | (256B align) | bf16 weights
  float* sums = (float*)d_ws;
  float* cnt  = sums + (size_t)N * 64;
  size_t woff = (((size_t)N * 65 * sizeof(float)) + 255) & ~(size_t)255;
  __bf16* wb  = (__bf16*)((char*)d_ws + woff);
  __bf16* w1s = wb;             // mw1: Kc=3, Nt=8 -> 12288 elems
  __bf16* w2s = wb + 12288;     // mw2: Kc=4, Nt=4 ->  8192
  __bf16* u1s = wb + 20480;     // uw1: Kc=5, Nt=8 -> 20480
  __bf16* u2s = wb + 40960;     // uw2: Kc=4, Nt=4 ->  8192

  int zn = N * 65;
  k_zero<<<(zn + 255) / 256, 256, 0, stream>>>(sums, zn);
  k_swz<<<(12288 + 255) / 256, 256, 0, stream>>>(mw1, w1s, 128, 8, 12288);
  k_swz<<<( 8192 + 255) / 256, 256, 0, stream>>>(mw2, w2s,  64, 4,  8192);
  k_swz<<<(20480 + 255) / 256, 256, 0, stream>>>(uw1, u1s, 128, 8, 20480);
  k_swz<<<( 8192 + 255) / 256, 256, 0, stream>>>(uw2, u2s,  64, 4,  8192);

  int etiles = (E + 31) / 32;
  k_edge<<<(etiles + 7) / 8, 256, 0, stream>>>(x, ei, ei + E, ea, wts, mb1, mb2,
                                               w1s, w2s, sums, cnt, E, etiles);
  int ntls = (N + 31) / 32;
  k_node<<<(ntls + 7) / 8, 256, 0, stream>>>(x, nb, u, sums, cnt, ub1, ub2,
                                             u1s, u2s, (float*)d_out, N, ntls);
}
